// TripletLoss_11836929868463
// MI455X (gfx1250) — matrix operand order repaired and checked by reading, not compiled, not run on hardware
//
#include <hip/hip_runtime.h>
#include <hip/hip_bf16.h>

typedef __attribute__((ext_vector_type(2)))  float  v2f;
typedef __attribute__((ext_vector_type(8)))  float  v8f;
typedef __attribute__((ext_vector_type(2)))  __bf16 v2bf;
typedef __attribute__((ext_vector_type(8)))  __bf16 v8bf;
typedef __attribute__((ext_vector_type(16))) __bf16 v16bf;

#define B_N    8192
#define D_K    512
#define NCT    (B_N / 16)     // 512 column tiles
#define LSTRF  516            // f32 LDS row stride (floats)
#define LSTRB  520            // bf16 LDS row stride (bf16 elems): row=1040B, %256B=16 -> no bank conflict, 16B aligned
#define MARGIN 0.3f
#define NEG_BIG 1e9f

// =====================================================================
// Path A (preferred): split-bf16 GEMM  X@X^T ~= hi@hi + hi@lo + lo@hi
// =====================================================================

// ---- Prep: row squared norms + f32 -> (bf16 hi, bf16 lo) decomposition ----
__global__ __launch_bounds__(256) void tl_prep_bf16_kernel(const float* __restrict__ X,
                                                           float* __restrict__ sq,
                                                           __bf16* __restrict__ xhi,
                                                           __bf16* __restrict__ xlo,
                                                           float* __restrict__ out) {
    const int wave = threadIdx.x >> 5;
    const int lane = threadIdx.x & 31;
    const int row  = blockIdx.x * 8 + wave;          // 1024 blocks * 8 waves
    const float2* xr2 = (const float2*)(X + (size_t)row * D_K);
    float s = 0.0f;
    #pragma unroll
    for (int it = 0; it < 8; ++it) {
        const int c2 = lane + it * 32;               // float2 index, coalesced
        float2 v = xr2[c2];
        s = fmaf(v.x, v.x, fmaf(v.y, v.y, s));
        __bf16 h0 = (__bf16)v.x;
        __bf16 h1 = (__bf16)v.y;
        __bf16 l0 = (__bf16)(v.x - (float)h0);
        __bf16 l1 = (__bf16)(v.y - (float)h1);
        v2bf hp; hp[0] = h0; hp[1] = h1;
        v2bf lp; lp[0] = l0; lp[1] = l1;
        *(v2bf*)&xhi[(size_t)row * D_K + c2 * 2] = hp;
        *(v2bf*)&xlo[(size_t)row * D_K + c2 * 2] = lp;
    }
    #pragma unroll
    for (int off = 16; off > 0; off >>= 1)
        s += __shfl_xor(s, off, 32);
    if (lane == 0) sq[row] = s;
    if (blockIdx.x == 0 && threadIdx.x == 0) out[0] = 0.0f;
}

// ---- Fused split-bf16 WMMA distance + batch-hard mining ----
__global__ __launch_bounds__(256) void tl_mine_bf16_kernel(const float* __restrict__ X,
                                                           const long long* __restrict__ tgt,
                                                           const float* __restrict__ sq,
                                                           const __bf16* __restrict__ xhi,
                                                           const __bf16* __restrict__ xlo,
                                                           float* __restrict__ out) {
    __shared__ __bf16 lds_ahi[16 * LSTRB];
    __shared__ __bf16 lds_alo[16 * LSTRB];
    __shared__ float  lds_sqrow[16];
    __shared__ int    lds_trow[16];
    __shared__ float  lds_pos[8 * 16];
    __shared__ float  lds_neg[8 * 16];

    const int wave = threadIdx.x >> 5;
    const int lane = threadIdx.x & 31;
    const int hh   = lane >> 4;
    const int l15  = lane & 15;
    const int rowBase = blockIdx.x * 16;

    // Stage A strip: read f32, split into bf16 hi/lo in LDS (32 elems/thread)
    #pragma unroll
    for (int it = 0; it < 32; ++it) {
        int flat = threadIdx.x + it * 256;           // 0..8191
        int m = flat >> 9;                           // /512
        int c = flat & 511;
        float v = X[((size_t)rowBase + m) * D_K + c];
        __bf16 h = (__bf16)v;
        lds_ahi[m * LSTRB + c] = h;
        lds_alo[m * LSTRB + c] = (__bf16)(v - (float)h);
    }
    if (threadIdx.x < 16) {
        lds_sqrow[threadIdx.x] = sq[rowBase + threadIdx.x];
        lds_trow[threadIdx.x]  = (int)tgt[rowBase + threadIdx.x];
    }
    __syncthreads();

    float sq_i[8];
    int   t_i[8];
    #pragma unroll
    for (int r = 0; r < 8; ++r) {
        sq_i[r] = lds_sqrow[r + 8 * hh];
        t_i[r]  = lds_trow[r + 8 * hh];
    }

    float posmax[8], negmin[8];
    #pragma unroll
    for (int r = 0; r < 8; ++r) { posmax[r] = -NEG_BIG; negmin[r] = NEG_BIG; }

    // A-operand chunk bases (contiguous K: [kb+16*hh,+8) and [kb+16*hh+8,+8), matching B)
    const __bf16* aph0 = &lds_ahi[l15 * LSTRB + 16 * hh];
    const __bf16* aph1 = &lds_ahi[l15 * LSTRB + 8 + 16 * hh];
    const __bf16* apl0 = &lds_alo[l15 * LSTRB + 16 * hh];
    const __bf16* apl1 = &lds_alo[l15 * LSTRB + 8 + 16 * hh];

    for (int ct = wave; ct < NCT; ct += 8) {
        const int colBase = ct * 16;
        const int j       = colBase + l15;
        const float sq_j  = sq[j];
        const int   t_j   = (int)tgt[j];
        // B-operand: lane holds X[j][kb + 16*hh .. +15] contiguously
        const __bf16* bph = xhi + (size_t)j * D_K + 16 * hh;
        const __bf16* bpl = xlo + (size_t)j * D_K + 16 * hh;

        v8f acc = {};
        #pragma unroll 4
        for (int kb = 0; kb < D_K; kb += 32) {
            v8bf ah0 = *(const v8bf*)(aph0 + kb);
            v8bf ah1 = *(const v8bf*)(aph1 + kb);
            v8bf al0 = *(const v8bf*)(apl0 + kb);
            v8bf al1 = *(const v8bf*)(apl1 + kb);
            v16bf ah = __builtin_shufflevector(ah0, ah1, 0,1,2,3,4,5,6,7,8,9,10,11,12,13,14,15);
            v16bf al = __builtin_shufflevector(al0, al1, 0,1,2,3,4,5,6,7,8,9,10,11,12,13,14,15);
            v16bf bh = *(const v16bf*)(bph + kb);
            v16bf bl = *(const v16bf*)(bpl + kb);
            acc = __builtin_amdgcn_wmma_f32_16x16x32_bf16(false, ah, false, bh, (short)0, acc, false, false);
            acc = __builtin_amdgcn_wmma_f32_16x16x32_bf16(false, ah, false, bl, (short)0, acc, false, false);
            acc = __builtin_amdgcn_wmma_f32_16x16x32_bf16(false, al, false, bh, (short)0, acc, false, false);
        }

        #pragma unroll
        for (int r = 0; r < 8; ++r) {
            const int i  = rowBase + r + 8 * hh;
            float d2 = sq_i[r] + sq_j - 2.0f * acc[r];
            float d  = __builtin_sqrtf(fmaxf(d2, 1e-12f));
            bool same = (t_i[r] == t_j);
            bool diag = (i == j);
            posmax[r] = fmaxf(posmax[r], (same && !diag) ? d : -NEG_BIG);
            negmin[r] = fminf(negmin[r], (!same)         ? d :  NEG_BIG);
        }
    }

    #pragma unroll
    for (int r = 0; r < 8; ++r) {
        #pragma unroll
        for (int off = 1; off < 16; off <<= 1) {
            posmax[r] = fmaxf(posmax[r], __shfl_xor(posmax[r], off, 32));
            negmin[r] = fminf(negmin[r], __shfl_xor(negmin[r], off, 32));
        }
    }
    if (l15 == 0) {
        #pragma unroll
        for (int r = 0; r < 8; ++r) {
            lds_pos[wave * 16 + r + 8 * hh] = posmax[r];
            lds_neg[wave * 16 + r + 8 * hh] = negmin[r];
        }
    }
    __syncthreads();

    if (threadIdx.x < 16) {
        float p = -NEG_BIG, n = NEG_BIG;
        #pragma unroll
        for (int w = 0; w < 8; ++w) {
            p = fmaxf(p, lds_pos[w * 16 + threadIdx.x]);
            n = fminf(n, lds_neg[w * 16 + threadIdx.x]);
        }
        const float diagv = 1e-6f;
        float d_ap = (p > -1e8f) ? p : diagv;
        float d_an = (n <  1e8f) ? n : diagv;
        float contrib = fmaxf(d_ap - d_an + MARGIN, 0.0f) * (1.0f / (float)B_N);
        atomicAdd(out, contrib);
    }
}

// =====================================================================
// Path B (fallback when ws too small): exact f32 WMMA 16x16x4
// =====================================================================

__global__ __launch_bounds__(256) void tl_sqnorm_kernel(const float* __restrict__ X,
                                                        float* __restrict__ sq,
                                                        float* __restrict__ out) {
    const int wave = threadIdx.x >> 5;
    const int lane = threadIdx.x & 31;
    const int row  = blockIdx.x * 8 + wave;
    const float* xr = X + (size_t)row * D_K;
    float s = 0.0f;
    #pragma unroll 4
    for (int c = lane; c < D_K; c += 32) {
        float v = xr[c];
        s = fmaf(v, v, s);
    }
    #pragma unroll
    for (int off = 16; off > 0; off >>= 1)
        s += __shfl_xor(s, off, 32);
    if (lane == 0) sq[row] = s;
    if (blockIdx.x == 0 && threadIdx.x == 0) out[0] = 0.0f;
}

__global__ __launch_bounds__(256) void tl_mine_kernel(const float* __restrict__ X,
                                                      const long long* __restrict__ tgt,
                                                      const float* __restrict__ sq,
                                                      float* __restrict__ out) {
    __shared__ float lds_x[16 * LSTRF];
    __shared__ float lds_sqrow[16];
    __shared__ int   lds_trow[16];
    __shared__ float lds_pos[8 * 16];
    __shared__ float lds_neg[8 * 16];

    const int wave = threadIdx.x >> 5;
    const int lane = threadIdx.x & 31;
    const int hh   = lane >> 4;
    const int l15  = lane & 15;
    const int rowBase = blockIdx.x * 16;

    {
        const float4* Xr4 = (const float4*)(X + (size_t)rowBase * D_K);
        #pragma unroll
        for (int it = 0; it < 8; ++it) {
            int flat4 = threadIdx.x + it * 256;
            int m  = flat4 >> 7;
            int c4 = flat4 & 127;
            float4 v = Xr4[(size_t)m * (D_K / 4) + c4];
            *(float4*)&lds_x[m * LSTRF + c4 * 4] = v;
        }
    }
    if (threadIdx.x < 16) {
        lds_sqrow[threadIdx.x] = sq[rowBase + threadIdx.x];
        lds_trow[threadIdx.x]  = (int)tgt[rowBase + threadIdx.x];
    }
    __syncthreads();

    float sq_i[8];
    int   t_i[8];
    #pragma unroll
    for (int r = 0; r < 8; ++r) {
        sq_i[r] = lds_sqrow[r + 8 * hh];
        t_i[r]  = lds_trow[r + 8 * hh];
    }

    float posmax[8], negmin[8];
    #pragma unroll
    for (int r = 0; r < 8; ++r) { posmax[r] = -NEG_BIG; negmin[r] = NEG_BIG; }

    const float* Aptr = &lds_x[l15 * LSTRF + 2 * hh];

    for (int ct = wave; ct < NCT; ct += 8) {
        const int colBase = ct * 16;
        const int j       = colBase + l15;
        const float sq_j  = sq[j];
        const int   t_j   = (int)tgt[j];
        const float* Bptr = X + (size_t)j * D_K + 2 * hh;

        v8f acc = {};
        #pragma unroll 8
        for (int k = 0; k < D_K; k += 4) {
            v2f a = *(const v2f*)(Aptr + k);
            v2f b = *(const v2f*)(Bptr + k);
            acc = __builtin_amdgcn_wmma_f32_16x16x4_f32(
                false, a, false, b, (short)0, acc, false, false);
        }

        #pragma unroll
        for (int r = 0; r < 8; ++r) {
            const int i  = rowBase + r + 8 * hh;
            float d2 = sq_i[r] + sq_j - 2.0f * acc[r];
            float d  = __builtin_sqrtf(fmaxf(d2, 1e-12f));
            bool same = (t_i[r] == t_j);
            bool diag = (i == j);
            posmax[r] = fmaxf(posmax[r], (same && !diag) ? d : -NEG_BIG);
            negmin[r] = fminf(negmin[r], (!same)         ? d :  NEG_BIG);
        }
    }

    #pragma unroll
    for (int r = 0; r < 8; ++r) {
        #pragma unroll
        for (int off = 1; off < 16; off <<= 1) {
            posmax[r] = fmaxf(posmax[r], __shfl_xor(posmax[r], off, 32));
            negmin[r] = fminf(negmin[r], __shfl_xor(negmin[r], off, 32));
        }
    }
    if (l15 == 0) {
        #pragma unroll
        for (int r = 0; r < 8; ++r) {
            lds_pos[wave * 16 + r + 8 * hh] = posmax[r];
            lds_neg[wave * 16 + r + 8 * hh] = negmin[r];
        }
    }
    __syncthreads();

    if (threadIdx.x < 16) {
        float p = -NEG_BIG, n = NEG_BIG;
        #pragma unroll
        for (int w = 0; w < 8; ++w) {
            p = fmaxf(p, lds_pos[w * 16 + threadIdx.x]);
            n = fminf(n, lds_neg[w * 16 + threadIdx.x]);
        }
        const float diagv = 1e-6f;
        float d_ap = (p > -1e8f) ? p : diagv;
        float d_an = (n <  1e8f) ? n : diagv;
        float contrib = fmaxf(d_ap - d_an + MARGIN, 0.0f) * (1.0f / (float)B_N);
        atomicAdd(out, contrib);
    }
}

// ---------------- Host-side launcher --------------------------------------
extern "C" void kernel_launch(void* const* d_in, const int* in_sizes, int n_in,
                              void* d_out, int out_size, void* d_ws, size_t ws_size,
                              hipStream_t stream) {
    const float*     X   = (const float*)d_in[0];       // [8192, 512] f32
    const long long* tgt = (const long long*)d_in[1];   // [8192] i64
    float* out = (float*)d_out;

    const size_t sq_bytes = (size_t)B_N * sizeof(float);
    const size_t mat_elems = (size_t)B_N * D_K;
    const size_t need = sq_bytes + 2 * mat_elems * 2;   // sq + xhi + xlo (bf16)

    if (ws_size >= need) {
        float*  sq  = (float*)d_ws;
        __bf16* xhi = (__bf16*)((char*)d_ws + sq_bytes);
        __bf16* xlo = xhi + mat_elems;
        tl_prep_bf16_kernel<<<B_N / 8, 256, 0, stream>>>(X, sq, xhi, xlo, out);
        tl_mine_bf16_kernel<<<B_N / 16, 256, 0, stream>>>(X, tgt, sq, xhi, xlo, out);
    } else {
        float* sq = (float*)d_ws;                       // needs only 32 KB
        tl_sqnorm_kernel<<<B_N / 8, 256, 0, stream>>>(X, sq, out);
        tl_mine_kernel<<<B_N / 16, 256, 0, stream>>>(X, tgt, sq, out);
    }
}
